// Att_76381698392129
// MI455X (gfx1250) — matrix operand order repaired and checked by reading, not compiled, not run on hardware
//
#include <hip/hip_runtime.h>

#define N_     5023
#define HD     256
#define BATCH  64
#define NP     5120          // 40*128 = 160*32 padded dim
#define LDA    40            // LDS row stride in ushort (80B, keeps 16B alignment)
#define KCH    40            // k-steps per split-K chunk in GEMM2 (160/4)

typedef unsigned short ushort_t;
typedef __attribute__((ext_vector_type(16))) __bf16 v16bf;
typedef __attribute__((ext_vector_type(8)))  __bf16 v8bf;
typedef __attribute__((ext_vector_type(8)))  float  v8f;

static __device__ __forceinline__ unsigned short f2bf(float f) {
    unsigned u = __float_as_uint(f);
    unsigned r = u + 0x7fffu + ((u >> 16) & 1u);   // RNE
    return (unsigned short)(r >> 16);
}
static __device__ __forceinline__ float bf2f(unsigned short h) {
    return __uint_as_float(((unsigned)h) << 16);
}
static __device__ __forceinline__ v8f wmma_bf16(v16bf a, v16bf b, v8f c) {
    return __builtin_amdgcn_wmma_f32_16x16x32_bf16(false, a, false, b, (short)0, c, false, false);
}

// A-fragment (16-bit 16x32 ISA layout): elems 0..7 -> K = sel*8 + e ; 8..15 -> K = 16 + sel*8 + e
static __device__ __forceinline__ v16bf load_frag_a(const unsigned short* row, int sel) {
    union { v16bf v; v8bf h[2]; } f;
    f.h[0] = *(const v8bf*)(const void*)(row + sel * 8);
    f.h[1] = *(const v8bf*)(const void*)(row + sel * 8 + 16);
    return f.v;
}
// B-fragment (LDS staged column-major): lane = column n, elems 0..7 -> K = sel*8 + e ; 8..15 -> K = 16 + sel*8 + e
static __device__ __forceinline__ v16bf load_frag_b(const unsigned short* row, int sel) {
    union { v16bf v; v8bf h[2]; } f;
    f.h[0] = *(const v8bf*)(const void*)(row + sel * 8);
    f.h[1] = *(const v8bf*)(const void*)(row + sel * 8 + 16);
    return f.v;
}
static __device__ __forceinline__ uint4 pack8(v8f a, float s) {
    uint4 u;
    u.x = (unsigned)f2bf(a[0] * s) | ((unsigned)f2bf(a[1] * s) << 16);
    u.y = (unsigned)f2bf(a[2] * s) | ((unsigned)f2bf(a[3] * s) << 16);
    u.z = (unsigned)f2bf(a[4] * s) | ((unsigned)f2bf(a[5] * s) << 16);
    u.w = (unsigned)f2bf(a[6] * s) | ((unsigned)f2bf(a[7] * s) << 16);
    return u;
}

// ---- prep: fp32 -> bf16 hi/lo, zero-padded. key row-major [i][k]; query pre-transposed [o][k] --
__global__ __launch_bounds__(256) void prep_key(const float* __restrict__ key_w,
                                                unsigned short* __restrict__ khi,
                                                unsigned short* __restrict__ klo) {
    const int e = blockIdx.x * 256 + threadIdx.x;       // NP*HD elements
    const int i = e >> 8, k = e & 255;
    float v = (i < N_) ? key_w[i * HD + k] : 0.f;
    unsigned short h = f2bf(v);
    khi[e] = h;
    klo[e] = f2bf(v - bf2f(h));
}
__global__ __launch_bounds__(256) void prep_query(const float* __restrict__ query_w,
                                                  unsigned short* __restrict__ qhi,
                                                  unsigned short* __restrict__ qlo) {
    const int e = blockIdx.x * 256 + threadIdx.x;       // NP*HD elements, layout [o][k]
    const int o = e >> 8, k = e & 255;
    float v = (o < N_) ? query_w[k * N_ + o] : 0.f;
    unsigned short h = f2bf(v);
    qhi[e] = h;
    qlo[e] = f2bf(v - bf2f(h));
}

// ---- GEMM1: scoreT[o,i] = bf16( (1/16) * key[i,:] . q[:,o] ), 3xbf16 split, 128x128 blocks ----
__global__ __launch_bounds__(256) void gemm1_score(const unsigned short* __restrict__ khi,
                                                   const unsigned short* __restrict__ klo,
                                                   const unsigned short* __restrict__ qhi,
                                                   const unsigned short* __restrict__ qlo,
                                                   unsigned short* __restrict__ scoreT) {
    __shared__ __align__(16) unsigned short As_hi[128 * LDA], As_lo[128 * LDA];
    __shared__ __align__(16) unsigned short Bs_hi[128 * LDA], Bs_lo[128 * LDA];
    const int tid = threadIdx.x, lane = tid & 31, w = tid >> 5;
    const int bi = blockIdx.x, bj = blockIdx.y;
    const int ti0 = (w & 3) * 2;        // 2 row tiles
    const int tj0 = (w >> 2) * 4;       // 4 col tiles
    const int sel = lane >> 4, nn = lane & 15;

    v8f acc[2][4] = {};

    for (int kt = 0; kt < 8; ++kt) {
        const int k0 = kt * 32;
        __syncthreads();
        {   // stage A (key[i][k]) and B (qT[o][k]) tiles: pure uint4 copies, unguarded
            const int c = tid & 3;
#pragma unroll
            for (int it = 0; it < 2; ++it) {
                const int row = (tid >> 2) + it * 64;
                const size_t ga = (size_t)(bi * 128 + row) * HD + k0 + c * 8;
                const size_t gb = (size_t)(bj * 128 + row) * HD + k0 + c * 8;
                *(uint4*)(void*)&As_hi[row * LDA + c * 8] = *(const uint4*)(const void*)(khi + ga);
                *(uint4*)(void*)&As_lo[row * LDA + c * 8] = *(const uint4*)(const void*)(klo + ga);
                *(uint4*)(void*)&Bs_hi[row * LDA + c * 8] = *(const uint4*)(const void*)(qhi + gb);
                *(uint4*)(void*)&Bs_lo[row * LDA + c * 8] = *(const uint4*)(const void*)(qlo + gb);
            }
        }
        __syncthreads();

        v16bf ah0 = load_frag_a(&As_hi[((ti0 + 0) * 16 + nn) * LDA], sel);
        v16bf al0 = load_frag_a(&As_lo[((ti0 + 0) * 16 + nn) * LDA], sel);
        v16bf ah1 = load_frag_a(&As_hi[((ti0 + 1) * 16 + nn) * LDA], sel);
        v16bf al1 = load_frag_a(&As_lo[((ti0 + 1) * 16 + nn) * LDA], sel);
#pragma unroll
        for (int t = 0; t < 4; ++t) {
            v16bf bh = load_frag_b(&Bs_hi[((tj0 + t) * 16 + nn) * LDA], sel);
            v16bf bl = load_frag_b(&Bs_lo[((tj0 + t) * 16 + nn) * LDA], sel);
            acc[0][t] = wmma_bf16(ah0, bh, acc[0][t]);
            acc[0][t] = wmma_bf16(ah0, bl, acc[0][t]);
            acc[0][t] = wmma_bf16(al0, bh, acc[0][t]);
            acc[1][t] = wmma_bf16(ah1, bh, acc[1][t]);
            acc[1][t] = wmma_bf16(ah1, bl, acc[1][t]);
            acc[1][t] = wmma_bf16(al1, bh, acc[1][t]);
        }
    }
    // epilogue: transposed store -> each lane writes one contiguous uint4 (8 bf16 along i)
#pragma unroll
    for (int i2 = 0; i2 < 2; ++i2) {
        const int row0 = bi * 128 + (ti0 + i2) * 16 + sel * 8;
#pragma unroll
        for (int t = 0; t < 4; ++t) {
            const int col = bj * 128 + (tj0 + t) * 16 + nn;
            *(uint4*)(void*)&scoreT[(size_t)col * NP + row0] = pack8(acc[i2][t], 0.0625f);
        }
    }
}

// ---- L1 partial sums over o (columns of scoreT), coalesced; atomically accumulated ------------
__global__ __launch_bounds__(256) void l1_partial(const unsigned short* __restrict__ scoreT,
                                                  float* __restrict__ l1sum) {
    const int i = blockIdx.x * 256 + threadIdx.x;       // NP lanes of i
    const int o0 = blockIdx.y * (NP / 16);              // 320 o per block
    float s = 0.f;
    for (int o = o0; o < o0 + NP / 16; ++o) s += fabsf(bf2f(scoreT[(size_t)o * NP + i]));
    atomicAdd(&l1sum[i], s);
}

// ---- build V transposed: VpT[b*3+e, i] = bf16( (x[b,i,:] . value_w[e,:]) / max(l1,eps) ) ------
__global__ __launch_bounds__(256) void build_v(const float* __restrict__ x,
                                               const float* __restrict__ vw,
                                               const float* __restrict__ l1sum,
                                               unsigned short* __restrict__ VpT) {
    const int i = blockIdx.x * 256 + threadIdx.x;       // NP
    const int b = blockIdx.y;                           // BATCH
    unsigned short o0 = 0, o1 = 0, o2 = 0;
    if (i < N_) {
        const float* xp = x + ((size_t)b * N_ + i) * 3;
        const float x0 = xp[0], x1 = xp[1], x2 = xp[2];
        const float inv = 1.0f / fmaxf(l1sum[i], 1e-12f);
        o0 = f2bf((x0 * vw[0] + x1 * vw[1] + x2 * vw[2]) * inv);
        o1 = f2bf((x0 * vw[3] + x1 * vw[4] + x2 * vw[5]) * inv);
        o2 = f2bf((x0 * vw[6] + x1 * vw[7] + x2 * vw[8]) * inv);
    }
    VpT[(size_t)(b * 3 + 0) * NP + i] = o0;
    VpT[(size_t)(b * 3 + 1) * NP + i] = o1;
    VpT[(size_t)(b * 3 + 2) * NP + i] = o2;
}

// ---- GEMM2: out[b,o,e] += sum_i scoreT[o,i] * VpT[b*3+e,i]  (split-K x4, fp32 atomics) --------
__global__ __launch_bounds__(256) void gemm2_out(const unsigned short* __restrict__ scoreT,
                                                 const unsigned short* __restrict__ VpT,
                                                 float* __restrict__ out) {
    __shared__ __align__(16) unsigned short A2s[64 * LDA];     // [o_local][i_local]
    __shared__ __align__(16) unsigned short B2s[192 * LDA];    // [j][i_local]
    const int tid = threadIdx.x, lane = tid & 31, w = tid >> 5;
    const int ob = blockIdx.x, kb = blockIdx.y;
    const int ti0 = (w & 1) * 2;        // 2 row tiles
    const int tj0 = (w >> 1) * 3;       // 3 col tiles
    const int sel = lane >> 4, nn = lane & 15;

    const int ktot = NP / 32;           // 160
    int ks0 = kb * KCH;
    int ks1 = ks0 + KCH; if (ks1 > ktot) ks1 = ktot;

    v8f acc[2][3] = {};

    for (int ks = ks0; ks < ks1; ++ks) {
        const int i0 = ks * 32;
        __syncthreads();
        {   // A2s: 64 o-rows x 32 i, straight uint4 copy from transposed score
            const int row = tid >> 2, c = tid & 3;
            *(uint4*)(void*)&A2s[row * LDA + c * 8] =
                *(const uint4*)(const void*)(scoreT + (size_t)(ob * 64 + row) * NP + i0 + c * 8);
        }
#pragma unroll
        for (int it = 0; it < 3; ++it) {  // B2s: 192 j-rows x 32 i
            const int e = tid + it * 256;
            const int j = e >> 2, c = e & 3;
            *(uint4*)(void*)&B2s[j * LDA + c * 8] =
                *(const uint4*)(const void*)(VpT + (size_t)j * NP + i0 + c * 8);
        }
        __syncthreads();

        v16bf a0 = load_frag_a(&A2s[((ti0 + 0) * 16 + nn) * LDA], sel);
        v16bf a1 = load_frag_a(&A2s[((ti0 + 1) * 16 + nn) * LDA], sel);
#pragma unroll
        for (int t = 0; t < 3; ++t) {
            v16bf b = load_frag_b(&B2s[((tj0 + t) * 16 + nn) * LDA], sel);
            acc[0][t] = wmma_bf16(a0, b, acc[0][t]);
            acc[1][t] = wmma_bf16(a1, b, acc[1][t]);
        }
    }

#pragma unroll
    for (int i2 = 0; i2 < 2; ++i2) {
        const int obase = ob * 64 + (ti0 + i2) * 16 + sel * 8;
#pragma unroll
        for (int t = 0; t < 3; ++t) {
            const int j = (tj0 + t) * 16 + nn;          // < 192
            const int bb = j / 3, e = j - bb * 3;
#pragma unroll
            for (int r = 0; r < 8; ++r) {
                const int o = obase + r;
                if (o < N_) atomicAdd(&out[((size_t)bb * N_ + o) * 3 + e], acc[i2][t][r]);
            }
        }
    }
}

extern "C" void kernel_launch(void* const* d_in, const int* in_sizes, int n_in,
                              void* d_out, int out_size, void* d_ws, size_t ws_size,
                              hipStream_t stream) {
    const float* x       = (const float*)d_in[0];   // [64, 5023, 3]
    const float* key_w   = (const float*)d_in[1];   // [5023, 256]
    const float* query_w = (const float*)d_in[2];   // [256, 5023]
    const float* value_w = (const float*)d_in[3];   // [3, 3]
    float* out = (float*)d_out;                     // [64, 5023, 3]

    char* ws = (char*)d_ws;
    size_t off = 0;
    unsigned short* scoreT = (unsigned short*)(ws + off); off += (size_t)NP * NP * 2;    // 52.4 MB
    unsigned short* khi    = (unsigned short*)(ws + off); off += (size_t)NP * HD * 2;
    unsigned short* klo    = (unsigned short*)(ws + off); off += (size_t)NP * HD * 2;
    unsigned short* qhi    = (unsigned short*)(ws + off); off += (size_t)NP * HD * 2;
    unsigned short* qlo    = (unsigned short*)(ws + off); off += (size_t)NP * HD * 2;
    unsigned short* VpT    = (unsigned short*)(ws + off); off += (size_t)192 * NP * 2;
    float*          l1sum  = (float*)(ws + off);          off += (size_t)NP * 4;

    hipMemsetAsync(d_out, 0, (size_t)out_size * sizeof(float), stream);
    hipMemsetAsync(l1sum, 0, (size_t)NP * sizeof(float), stream);

    prep_key  <<<(NP * HD) / 256, 256, 0, stream>>>(key_w, khi, klo);
    prep_query<<<(NP * HD) / 256, 256, 0, stream>>>(query_w, qhi, qlo);
    gemm1_score<<<dim3(NP / 128, NP / 128), 256, 0, stream>>>(khi, klo, qhi, qlo, scoreT);
    l1_partial<<<dim3(NP / 256, 16), 256, 0, stream>>>(scoreT, l1sum);
    build_v<<<dim3(NP / 256, BATCH), 256, 0, stream>>>(x, value_w, l1sum, VpT);
    gemm2_out<<<dim3(NP / 64, 4), 256, 0, stream>>>(scoreT, VpT, out);
}
